// GCN_9552007266308
// MI455X (gfx1250) — matrix operand order repaired and checked by reading, not compiled, not run on hardware
//
#include <hip/hip_runtime.h>
#include <hip/hip_bf16.h>

typedef _Float16 h16 __attribute__((ext_vector_type(16)));
typedef float    f8  __attribute__((ext_vector_type(8)));

#define NN 50000
#define NE 800000
#define NG 64
#define C0 32
#define C1 64
#define C2 128

// ---------------------------------------------------------------- utilities
__global__ void fill_kernel(float* __restrict__ p, float v, int n) {
    int i = blockIdx.x * blockDim.x + threadIdx.x;
    if (i < n) p[i] = v;
}

__global__ void deg_kernel(const int* __restrict__ dst, float* __restrict__ deg) {
    int e = blockIdx.x * blockDim.x + threadIdx.x;
    if (e < NE) atomicAdd(&deg[dst[e]], 1.0f);
}

__global__ void dinv_kernel(float* __restrict__ deg) {
    int i = blockIdx.x * blockDim.x + threadIdx.x;
    if (i < NN) deg[i] = rsqrtf(deg[i] + 1.0f);   // deg>=0, +1 self loop => never 0
}

// ------------------------------------------------------- WMMA GEMM layer 1
// HS = (X @ W1) * dinv[row] ; X fp32 [NN,32], W1 fp32 [32,64], HS fp32 [NN,64]
__global__ void gemm1_wmma(const float* __restrict__ X, const float* __restrict__ W,
                           const float* __restrict__ dinv, float* __restrict__ HS) {
    const int lane = threadIdx.x & 31;
    const int wave = threadIdx.x >> 5;
    const int NT = C1 / 16;          // 4 N-tiles
    const int MT = NN / 16;          // 3125 M-tiles (exact)
    int gw = blockIdx.x * 8 + wave;
    if (gw >= MT * NT) return;       // wave-uniform: EXEC all-ones inside
    int mt = gw / NT, nt = gw % NT;

    // A fragment: 16x32 f16.  lanes0-15: K 0..7 & 16..23 ; lanes16-31: K 8..15 & 24..31
    int row = mt * 16 + (lane & 15);
    int klo = (lane >> 4) * 8;
    const float* xr = X + (size_t)row * C0;
    h16 a;
#pragma unroll
    for (int e = 0; e < 8; ++e) a[e]     = (_Float16)xr[klo + e];
#pragma unroll
    for (int e = 0; e < 8; ++e) a[8 + e] = (_Float16)xr[16 + klo + e];

    // B fragment: 32x16 f16. lanes0-15: K 0..7 & 16..23 ; lanes16-31: K 8..15 & 24..31 (col = lane%16)
    int col  = nt * 16 + (lane & 15);
    int kofs = (lane >> 4) * 16;
    h16 b;
#pragma unroll
    for (int e = 0; e < 16; ++e) b[e] = (_Float16)W[(size_t)((e < 8) ? (klo + e) : (16 + klo + e - 8)) * C1 + col];

    f8 c = {};
    c = __builtin_amdgcn_wmma_f32_16x16x32_f16(false, a, false, b, (short)0, c, false, false);

    // C/D layout: VGPR r -> M = r (lanes0-15) / 8+r (lanes16-31), N = lane%16
    int mrow = mt * 16 + ((lane >> 4) << 3);
#pragma unroll
    for (int r = 0; r < 8; ++r) {
        int m = mrow + r;
        HS[(size_t)m * C1 + col] = c[r] * dinv[m];
    }
}

// ------------------------------------------------------- WMMA GEMM layer 2
// HS2 = (H @ W2) * dinv[row] ; H f16 [NN,64], W2 fp32 [64,128]
__global__ void gemm2_wmma(const _Float16* __restrict__ H, const float* __restrict__ W,
                           const float* __restrict__ dinv, float* __restrict__ HS) {
    const int lane = threadIdx.x & 31;
    const int wave = threadIdx.x >> 5;
    const int NT = C2 / 16;          // 8
    const int MT = NN / 16;          // 3125
    int gw = blockIdx.x * 8 + wave;
    if (gw >= MT * NT) return;
    int mt = gw / NT, nt = gw % NT;

    int row  = mt * 16 + (lane & 15);
    int klo  = (lane >> 4) * 8;
    int col  = nt * 16 + (lane & 15);
    int kofs = (lane >> 4) * 16;
    const _Float16* hr = H + (size_t)row * C1;

    f8 c = {};
#pragma unroll
    for (int kt = 0; kt < C1; kt += 32) {
        h16 a;
#pragma unroll
        for (int e = 0; e < 8; ++e) a[e]     = hr[kt + klo + e];
#pragma unroll
        for (int e = 0; e < 8; ++e) a[8 + e] = hr[kt + 16 + klo + e];
        h16 b;
#pragma unroll
        for (int e = 0; e < 16; ++e) b[e] = (_Float16)W[(size_t)(kt + ((e < 8) ? (klo + e) : (16 + klo + e - 8))) * C2 + col];
        c = __builtin_amdgcn_wmma_f32_16x16x32_f16(false, a, false, b, (short)0, c, false, false);
    }

    int mrow = mt * 16 + ((lane >> 4) << 3);
#pragma unroll
    for (int r = 0; r < 8; ++r) {
        int m = mrow + r;
        HS[(size_t)m * C2 + col] = c[r] * dinv[m];
    }
}

// ----------------------------------------------- edge scatter (L2-resident)
__global__ void edge_kernel64(const int* __restrict__ src, const int* __restrict__ dst,
                              const float* __restrict__ hs, float* __restrict__ agg) {
    int idx = blockIdx.x * blockDim.x + threadIdx.x;     // NE * 16 threads
    if (idx >= NE * 16) return;
    int e = idx >> 4, g = (idx & 15) << 2;
    int s = src[e], d = dst[e];
    const float4 v = *reinterpret_cast<const float4*>(hs + (size_t)s * C1 + g);
    float* ap = agg + (size_t)d * C1 + g;
    atomicAdd(ap + 0, v.x); atomicAdd(ap + 1, v.y);
    atomicAdd(ap + 2, v.z); atomicAdd(ap + 3, v.w);
}

__global__ void edge_kernel128(const int* __restrict__ src, const int* __restrict__ dst,
                               const float* __restrict__ hs, float* __restrict__ agg) {
    int idx = blockIdx.x * blockDim.x + threadIdx.x;     // NE * 32 threads
    if (idx >= NE * 32) return;
    int e = idx >> 5, g = (idx & 31) << 2;
    int s = src[e], d = dst[e];
    const float4 v = *reinterpret_cast<const float4*>(hs + (size_t)s * C2 + g);
    float* ap = agg + (size_t)d * C2 + g;
    atomicAdd(ap + 0, v.x); atomicAdd(ap + 1, v.y);
    atomicAdd(ap + 2, v.z); atomicAdd(ap + 3, v.w);
}

// ------------------------------------------------------------- epilogues
// out = relu(dinv*(agg + hs) + b); emit f16 for next GEMM's A operand
__global__ void fin1(const float* __restrict__ agg, const float* __restrict__ hs,
                     const float* __restrict__ dinv, const float* __restrict__ b1,
                     _Float16* __restrict__ out) {
    int idx = blockIdx.x * blockDim.x + threadIdx.x;
    if (idx >= NN * C1) return;
    int i = idx >> 6, c = idx & 63;
    float v = dinv[i] * (agg[idx] + hs[idx]) + b1[c];
    out[idx] = (_Float16)fmaxf(v, 0.0f);
}

// finalize layer2 fused with segment-max pooling (post-ReLU >= 0 => int atomicMax valid)
__global__ void fin2_pool(const float* __restrict__ agg, const float* __restrict__ hs,
                          const float* __restrict__ dinv, const float* __restrict__ b2,
                          const int* __restrict__ batch, float* __restrict__ pooled) {
    int idx = blockIdx.x * blockDim.x + threadIdx.x;
    if (idx >= NN * C2) return;
    int i = idx >> 7, c = idx & 127;
    float v = fmaxf(dinv[i] * (agg[idx] + hs[idx]) + b2[c], 0.0f);
    atomicMax((int*)&pooled[(size_t)batch[i] * C2 + c], __float_as_int(v));
}

// ------------------------------------------------------------- MLP head
__global__ void head_kernel(const float* __restrict__ pooled,
                            const float* __restrict__ W3, const float* __restrict__ b3,
                            const float* __restrict__ g3, const float* __restrict__ be3,
                            const float* __restrict__ W4, const float* __restrict__ b4,
                            const float* __restrict__ g4, const float* __restrict__ be4,
                            const float* __restrict__ W5, const float* __restrict__ b5,
                            float* __restrict__ out) {
    __shared__ float p[C2], z3[32], z4[64], lg[10], red[2];
    int g = blockIdx.x, t = threadIdx.x;
    if (t < C2) p[t] = pooled[(size_t)g * C2 + t];
    __syncthreads();
    const float bnc = rsqrtf(1.0f + 1e-5f);
    if (t < 32) {
        float acc = b3[t];
        for (int k = 0; k < C2; ++k) acc += p[k] * W3[k * 32 + t];
        acc = fmaxf(acc, 0.0f);
        z3[t] = acc * (g3[t] * bnc) + be3[t];
    }
    __syncthreads();
    if (t < 64) {
        float acc = b4[t];
        for (int k = 0; k < 32; ++k) acc += z3[k] * W4[k * 64 + t];
        acc = fmaxf(acc, 0.0f);
        z4[t] = acc * (g4[t] * bnc) + be4[t];
    }
    __syncthreads();
    if (t < 10) {
        float acc = b5[t];
        for (int k = 0; k < 64; ++k) acc += z4[k] * W5[k * 10 + t];
        lg[t] = acc;
    }
    __syncthreads();
    if (t == 0) {
        float m = lg[0];
        for (int j = 1; j < 10; ++j) m = fmaxf(m, lg[j]);
        float s = 0.0f;
        for (int j = 0; j < 10; ++j) s += expf(lg[j] - m);
        red[0] = m; red[1] = logf(s);
    }
    __syncthreads();
    if (t < 10) out[g * 10 + t] = lg[t] - red[0] - red[1];
}

// ---------------------------------------------------------------- driver
extern "C" void kernel_launch(void* const* d_in, const int* in_sizes, int n_in,
                              void* d_out, int out_size, void* d_ws, size_t ws_size,
                              hipStream_t stream) {
    (void)in_sizes; (void)n_in; (void)out_size; (void)ws_size;
    const float* x    = (const float*)d_in[0];
    const int*   esrc = (const int*)  d_in[1];
    const int*   edst = (const int*)  d_in[2];
    const int*   batch= (const int*)  d_in[3];
    const float* W1 = (const float*)d_in[4];  const float* b1  = (const float*)d_in[5];
    const float* W2 = (const float*)d_in[6];  const float* b2  = (const float*)d_in[7];
    const float* W3 = (const float*)d_in[8];  const float* b3  = (const float*)d_in[9];
    const float* g3 = (const float*)d_in[10]; const float* be3 = (const float*)d_in[11];
    const float* W4 = (const float*)d_in[12]; const float* b4  = (const float*)d_in[13];
    const float* g4 = (const float*)d_in[14]; const float* be4 = (const float*)d_in[15];
    const float* W5 = (const float*)d_in[16]; const float* b5  = (const float*)d_in[17];
    float* out = (float*)d_out;

    float* ws = (float*)d_ws;
    size_t off = 0;
    auto carve = [&](size_t n) { float* p = ws + off; off += (n + 255) & ~(size_t)255; return p; };
    float* deg    = carve(NN);                    // becomes dinv in place
    float* hs1    = carve((size_t)NN * C1);
    float* agg1   = carve((size_t)NN * C1);
    float* h1h_f  = carve((size_t)NN * C1 / 2);   // f16 storage
    float* hs2    = carve((size_t)NN * C2);
    float* agg2   = carve((size_t)NN * C2);
    float* pooled = carve((size_t)NG * C2);
    _Float16* h1h = (_Float16*)h1h_f;

    auto nb = [](long long n, int b) { return (unsigned)((n + b - 1) / b); };

    // zero accumulators every call (harness does not re-poison between replays)
    fill_kernel<<<nb(NN, 256), 256, 0, stream>>>(deg, 0.0f, NN);
    fill_kernel<<<nb((long long)NN * C1, 256), 256, 0, stream>>>(agg1, 0.0f, NN * C1);
    fill_kernel<<<nb((long long)NN * C2, 256), 256, 0, stream>>>(agg2, 0.0f, NN * C2);
    fill_kernel<<<nb(NG * C2, 256), 256, 0, stream>>>(pooled, 0.0f, NG * C2);

    deg_kernel <<<nb(NE, 256), 256, 0, stream>>>(edst, deg);
    dinv_kernel<<<nb(NN, 256), 256, 0, stream>>>(deg);

    // layer 1
    gemm1_wmma<<<nb((NN / 16) * (C1 / 16), 8), 256, 0, stream>>>(x, W1, deg, hs1);
    edge_kernel64<<<nb((long long)NE * 16, 256), 256, 0, stream>>>(esrc, edst, hs1, agg1);
    fin1<<<nb((long long)NN * C1, 256), 256, 0, stream>>>(agg1, hs1, deg, b1, h1h);

    // layer 2
    gemm2_wmma<<<nb((NN / 16) * (C2 / 16), 8), 256, 0, stream>>>(h1h, W2, deg, hs2);
    edge_kernel128<<<nb((long long)NE * 32, 256), 256, 0, stream>>>(esrc, edst, hs2, agg2);
    fin2_pool<<<nb((long long)NN * C2, 256), 256, 0, stream>>>(agg2, hs2, deg, b2, batch, pooled);

    // head
    head_kernel<<<NG, 128, 0, stream>>>(pooled, W3, b3, g3, be3, W4, b4, g4, be4, W5, b5, out);
}
